// CasualSelfAttention_24936580120617
// MI455X (gfx1250) — matrix operand order repaired and checked by reading, not compiled, not run on hardware
//
#include <hip/hip_runtime.h>
#include <hip/hip_bf16.h>
#include <math.h>

// ---------------------------------------------------------------------------
// Problem constants (from the reference)
// ---------------------------------------------------------------------------
#define DMODEL 1024
#define DKEY   64
#define NHEADS 16
#define BB     4
#define TT     2048
#define MROWS  (BB * TT)      // 8192 token rows

typedef __attribute__((ext_vector_type(16))) __bf16 v16bf;
typedef __attribute__((ext_vector_type(8)))  __bf16 v8bf;
typedef __attribute__((ext_vector_type(8)))  float  v8f;

// D = A(16x32 bf16) x B(32x16 bf16) + C(16x16 f32)
__device__ __forceinline__ v8f wmma_bf16(v16bf a, v16bf b, v8f c) {
    return __builtin_amdgcn_wmma_f32_16x16x32_bf16(
        /*neg_a=*/false, a, /*neg_b=*/false, b,
        /*c_mod=*/(short)0, c, /*reuse_a=*/false, /*reuse_b=*/false);
}

// 16-lane-group butterfly exchange via v_permlane16_b32 (pure VALU, no LDS).
__device__ __forceinline__ float permlane16_f(float x, unsigned lo, unsigned hi) {
    union { float f; unsigned u; } in, out;
    in.f = x;
    out.u = __builtin_amdgcn_permlane16(in.u, in.u, lo, hi, false, false);
    return out.f;
}
// XOR-butterfly selector pairs (nibble i = source lane for lane i within 16).
#define SEL1_LO 0x67452301u
#define SEL1_HI 0xEFCDAB89u
#define SEL2_LO 0x54761032u
#define SEL2_HI 0xDCFE98BAu
#define SEL4_LO 0x32107654u
#define SEL4_HI 0xBA98FEDCu
#define SEL8_LO 0xFEDCBA98u
#define SEL8_HI 0x76543210u

// ---------------------------------------------------------------------------
// Fragment loaders.
//  A-frag (16x32): lane l holds row m=l&15; element i -> K = (i<8?i:i+8)+8*half
//  B-frag (32x16): lane l holds col n=l&15; element i -> K = (i<8?i:i+8)+8*half
// ---------------------------------------------------------------------------
__device__ __forceinline__ v16bf fragA_f32(const float* p, int half) {
    const float4* c1 = (const float4*)(p + half * 8);
    const float4* c2 = (const float4*)(p + 16 + half * 8);
    float4 x0 = c1[0], x1 = c1[1], y0 = c2[0], y1 = c2[1];
    v16bf f;
    f[0]=(__bf16)x0.x;  f[1]=(__bf16)x0.y;  f[2]=(__bf16)x0.z;  f[3]=(__bf16)x0.w;
    f[4]=(__bf16)x1.x;  f[5]=(__bf16)x1.y;  f[6]=(__bf16)x1.z;  f[7]=(__bf16)x1.w;
    f[8]=(__bf16)y0.x;  f[9]=(__bf16)y0.y;  f[10]=(__bf16)y0.z; f[11]=(__bf16)y0.w;
    f[12]=(__bf16)y1.x; f[13]=(__bf16)y1.y; f[14]=(__bf16)y1.z; f[15]=(__bf16)y1.w;
    return f;
}

__device__ __forceinline__ v16bf fragA_bf16(const __bf16* p, int half) {
    v8bf c1 = *(const v8bf*)(p + half * 8);
    v8bf c2 = *(const v8bf*)(p + 16 + half * 8);
    v16bf f;
#pragma unroll
    for (int i = 0; i < 8; ++i) { f[i] = c1[i]; f[8 + i] = c2[i]; }
    return f;
}

__device__ __forceinline__ v16bf fragB_bf16(const __bf16* p, int half) {
    v8bf c1 = *(const v8bf*)(p + half * 8);
    v8bf c2 = *(const v8bf*)(p + 16 + half * 8);
    v16bf f;
#pragma unroll
    for (int i = 0; i < 8; ++i) { f[i] = c1[i]; f[8 + i] = c2[i]; }
    return f;
}

// ---------------------------------------------------------------------------
// f32 -> bf16 elementwise conversion (for the weight matrices, once/launch).
// ---------------------------------------------------------------------------
__global__ __launch_bounds__(256)
void cvt_f32_bf16_kernel(const float* __restrict__ in, __bf16* __restrict__ out) {
    const size_t i = ((size_t)blockIdx.x * 256 + threadIdx.x) * 8;
    float4 a = *(const float4*)(in + i);
    float4 b = *(const float4*)(in + i + 4);
    v8bf o;
    o[0]=(__bf16)a.x; o[1]=(__bf16)a.y; o[2]=(__bf16)a.z; o[3]=(__bf16)a.w;
    o[4]=(__bf16)b.x; o[5]=(__bf16)b.y; o[6]=(__bf16)b.z; o[7]=(__bf16)b.w;
    *(v8bf*)(out + i) = o;
}

// ---------------------------------------------------------------------------
// GEMM: out[m,n] = sum_k A[m,k]*W[n,k] + bias[n]
//   A: [8192,1024] f32 row-major (converted per-fragment), W: [1024,1024] bf16
//   MODE 0: bf16 out at [B,H,T,64]   (attention Q/K operands)
//   MODE 1: bf16 out at [B,H,64,T]   (transposed V for the P*V WMMA)
//   MODE 2: f32  out at [M,1024]     (final output projection)
// 256 threads = 8 waves; wave w computes rows m0..m0+15 x 128 cols (8 accs).
// ---------------------------------------------------------------------------
template <int MODE>
__global__ __launch_bounds__(256)
void gemm_wmma_kernel(const float* __restrict__ A, const __bf16* __restrict__ W,
                      const float* __restrict__ bias, void* __restrict__ out) {
    const int lane = threadIdx.x & 31;
    const int wave = threadIdx.x >> 5;
    const int half = lane >> 4;
    const int lcol = lane & 15;
    const int m0 = blockIdx.x * 128 + wave * 16;
    const int n0 = blockIdx.y * 128;

    v8f acc[8];
#pragma unroll
    for (int f = 0; f < 8; ++f) acc[f] = {};

    const float* arow = A + (size_t)(m0 + lcol) * DMODEL;
    for (int k0 = 0; k0 < DMODEL; k0 += 32) {
        __builtin_prefetch(arow + k0 + 128, 0, 3);
        v16bf a = fragA_f32(arow + k0, half);
#pragma unroll
        for (int f = 0; f < 8; ++f) {
            const __bf16* wrow = W + (size_t)(n0 + f * 16 + lcol) * DMODEL + k0;
            acc[f] = wmma_bf16(a, fragB_bf16(wrow, half), acc[f]);
        }
    }

#pragma unroll
    for (int f = 0; f < 8; ++f) {
        const int n = n0 + f * 16 + lcol;
        const float bv = bias[n];
#pragma unroll
        for (int v = 0; v < 8; ++v) {
            const float val = acc[f][v] + bv;
            const int m = m0 + v + 8 * half;
            if (MODE == 2) {
                ((float*)out)[(size_t)m * DMODEL + n] = val;
            } else {
                const int b = m >> 11, t = m & (TT - 1);
                const int h = n >> 6, d = n & (DKEY - 1);
                const size_t idx = (MODE == 0)
                    ? ((size_t)(b * NHEADS + h) * TT + t) * DKEY + d
                    : ((size_t)(b * NHEADS + h) * DKEY + d) * TT + t;
                ((__bf16*)out)[idx] = (__bf16)val;
            }
        }
    }
}

// ---------------------------------------------------------------------------
// Flash attention, one (b,h) per blockIdx.y, 64 query rows per block.
//   ctx[t] = softmax_s( Qa[t].Ka[s] / 8 ) . V[s]        (exp2-domain softmax)
//   Qa,Ka: [B*H, T, 64] bf16 ;  Vt: [B*H, 64, T] bf16 ; ctx: [B*T, 1024] f32
// Each of 4 waves owns 16 rows; streams keys in 64-wide tiles (16 WMMAs per
// softmax pass). P goes through a padded per-wave LDS tile (C->A layout).
// ---------------------------------------------------------------------------
__global__ __launch_bounds__(128)
void flash_attn_kernel(const __bf16* __restrict__ Qa, const __bf16* __restrict__ Ka,
                       const __bf16* __restrict__ Vt, float* __restrict__ ctx) {
    __shared__ __align__(16) __bf16 pstage[4][16][72];   // 64 cols + pad 8

    const int lane = threadIdx.x & 31;
    const int wave = threadIdx.x >> 5;
    const int half = lane >> 4;
    const int lcol = lane & 15;
    const int bh = blockIdx.y;
    const int t0 = blockIdx.x * 64 + wave * 16;
    const float c = 0.125f * 1.44269504088896340736f;    // 1/sqrt(dk) * log2(e)

    // Query A-fragments (16 rows x dkey=64 -> two K=32 fragments), loaded once.
    const __bf16* qrow = Qa + ((size_t)bh * TT + t0 + lcol) * DKEY;
    const v16bf aq0 = fragA_bf16(qrow, half);
    const v16bf aq1 = fragA_bf16(qrow + 32, half);

    v8f acc[4];
#pragma unroll
    for (int f = 0; f < 4; ++f) acc[f] = {};
    float mrow[8], lrow[8];
#pragma unroll
    for (int v = 0; v < 8; ++v) { mrow[v] = -INFINITY; lrow[v] = 0.0f; }

    const __bf16* kbase = Ka + ((size_t)bh * TT + lcol) * DKEY;
    const __bf16* vbase = Vt + ((size_t)bh * DKEY + lcol) * TT;

    for (int s0 = 0; s0 < TT; s0 += 64) {
        if (s0 + 64 < TT) {   // uniform branch: prefetch next K/V tiles
            __builtin_prefetch(kbase + (size_t)(s0 + 64) * DKEY, 0, 3);
            __builtin_prefetch(vbase + s0 + 64, 0, 3);
        }

        // S tile (16 rows x 64 keys, raw dot products) as four C fragments.
        v8f S[4];
#pragma unroll
        for (int j = 0; j < 4; ++j) {
            S[j] = {};
            const __bf16* krow = kbase + (size_t)(s0 + 16 * j) * DKEY;
            S[j] = wmma_bf16(aq0, fragB_bf16(krow, half), S[j]);
            S[j] = wmma_bf16(aq1, fragB_bf16(krow + 32, half), S[j]);
        }

        // Row max of raw scores across the 16 lanes of each half-wave.
        float mx[8];
#pragma unroll
        for (int v = 0; v < 8; ++v)
            mx[v] = fmaxf(fmaxf(S[0][v], S[1][v]), fmaxf(S[2][v], S[3][v]));
#pragma unroll
        for (int v = 0; v < 8; ++v) {
            mx[v] = fmaxf(mx[v], permlane16_f(mx[v], SEL1_LO, SEL1_HI));
            mx[v] = fmaxf(mx[v], permlane16_f(mx[v], SEL2_LO, SEL2_HI));
            mx[v] = fmaxf(mx[v], permlane16_f(mx[v], SEL4_LO, SEL4_HI));
            mx[v] = fmaxf(mx[v], permlane16_f(mx[v], SEL8_LO, SEL8_HI));
        }

        // Online rescale in exp2 domain: P = 2^(c*(S - mnew)).
        float alpha[8], mc[8];
#pragma unroll
        for (int v = 0; v < 8; ++v) {
            const float mn = fmaxf(mrow[v], mx[v]);
            alpha[v] = exp2f((mrow[v] - mn) * c);
            mrow[v] = mn;
            mc[v] = mn * c;
        }
        float rsum[8];
#pragma unroll
        for (int v = 0; v < 8; ++v) rsum[v] = 0.0f;
#pragma unroll
        for (int j = 0; j < 4; ++j)
#pragma unroll
            for (int v = 0; v < 8; ++v) {
                const float p = exp2f(fmaf(S[j][v], c, -mc[v]));
                S[j][v] = p;
                rsum[v] += p;
            }
#pragma unroll
        for (int v = 0; v < 8; ++v) {
            rsum[v] += permlane16_f(rsum[v], SEL1_LO, SEL1_HI);
            rsum[v] += permlane16_f(rsum[v], SEL2_LO, SEL2_HI);
            rsum[v] += permlane16_f(rsum[v], SEL4_LO, SEL4_HI);
            rsum[v] += permlane16_f(rsum[v], SEL8_LO, SEL8_HI);
            lrow[v] = lrow[v] * alpha[v] + rsum[v];
        }
#pragma unroll
        for (int f = 0; f < 4; ++f)
#pragma unroll
            for (int v = 0; v < 8; ++v) acc[f][v] *= alpha[v];

        // Stage P (bf16) through per-wave LDS: C layout -> A layout.
#pragma unroll
        for (int j = 0; j < 4; ++j)
#pragma unroll
            for (int v = 0; v < 8; ++v)
                pstage[wave][v + 8 * half][16 * j + lcol] = (__bf16)S[j][v];
        const v16bf pa0 = fragA_bf16(&pstage[wave][lcol][0], half);
        const v16bf pa1 = fragA_bf16(&pstage[wave][lcol][32], half);

        // acc += P (16x64) x V (64x64), V consumed from transposed layout.
#pragma unroll
        for (int f = 0; f < 4; ++f) {
            const __bf16* vrow = vbase + (size_t)(f * 16) * TT + s0;
            acc[f] = wmma_bf16(pa0, fragB_bf16(vrow, half), acc[f]);
            acc[f] = wmma_bf16(pa1, fragB_bf16(vrow + 32, half), acc[f]);
        }
    }

    // Normalize and store to ctx[B*T, H*64] (16 contiguous floats/half-wave).
    const int b = bh >> 4, h = bh & (NHEADS - 1);
    float rinv[8];
#pragma unroll
    for (int v = 0; v < 8; ++v) rinv[v] = 1.0f / lrow[v];
#pragma unroll
    for (int f = 0; f < 4; ++f)
#pragma unroll
        for (int v = 0; v < 8; ++v) {
            const int t = t0 + v + 8 * half;
            ctx[((size_t)(b * TT + t)) * DMODEL + h * DKEY + f * 16 + lcol] =
                acc[f][v] * rinv[v];
        }
}

// ---------------------------------------------------------------------------
// Launch: weight cvt -> 3 projection GEMMs -> flash attention -> output GEMM.
// Reference swaps K/Q in the score einsum: attention-query = keys@WK,
// attention-key = queries@WQ. No masking (reference discards masked_fill).
// Workspace: 4x2MB bf16 weights + Qa/Ka/Vt (3x16MB) + ctx (32MB) = 88MB.
// ---------------------------------------------------------------------------
extern "C" void kernel_launch(void* const* d_in, const int* in_sizes, int n_in,
                              void* d_out, int out_size, void* d_ws, size_t ws_size,
                              hipStream_t stream) {
    (void)in_sizes; (void)n_in; (void)out_size; (void)ws_size;
    const float* keys    = (const float*)d_in[0];
    const float* queries = (const float*)d_in[1];
    const float* values  = (const float*)d_in[2];
    const float* WKw = (const float*)d_in[4];
    const float* WKb = (const float*)d_in[5];
    const float* WQw = (const float*)d_in[6];
    const float* WQb = (const float*)d_in[7];
    const float* WVw = (const float*)d_in[8];
    const float* WVb = (const float*)d_in[9];
    const float* WOw = (const float*)d_in[10];
    const float* WOb = (const float*)d_in[11];

    char* ws = (char*)d_ws;
    const size_t wBytes    = (size_t)DMODEL * DMODEL * sizeof(__bf16);          // 2 MB
    const size_t projBytes = (size_t)BB * NHEADS * TT * DKEY * sizeof(__bf16);  // 16 MB
    __bf16* WbK = (__bf16*)(ws);
    __bf16* WbQ = (__bf16*)(ws + 1 * wBytes);
    __bf16* WbV = (__bf16*)(ws + 2 * wBytes);
    __bf16* WbO = (__bf16*)(ws + 3 * wBytes);
    __bf16* Qa  = (__bf16*)(ws + 4 * wBytes);
    __bf16* Ka  = (__bf16*)(ws + 4 * wBytes + projBytes);
    __bf16* Vt  = (__bf16*)(ws + 4 * wBytes + 2 * projBytes);
    float*  ctx = (float*)(ws + 4 * wBytes + 3 * projBytes);

    const int cvtGrid = (DMODEL * DMODEL) / (256 * 8);   // 512 blocks
    cvt_f32_bf16_kernel<<<cvtGrid, 256, 0, stream>>>(WKw, WbK);
    cvt_f32_bf16_kernel<<<cvtGrid, 256, 0, stream>>>(WQw, WbQ);
    cvt_f32_bf16_kernel<<<cvtGrid, 256, 0, stream>>>(WVw, WbV);
    cvt_f32_bf16_kernel<<<cvtGrid, 256, 0, stream>>>(WOw, WbO);

    dim3 ggrid(MROWS / 128, DMODEL / 128);
    gemm_wmma_kernel<0><<<ggrid, 256, 0, stream>>>(keys,    WbK, WKb, (void*)Qa);
    gemm_wmma_kernel<0><<<ggrid, 256, 0, stream>>>(queries, WbQ, WQb, (void*)Ka);
    gemm_wmma_kernel<1><<<ggrid, 256, 0, stream>>>(values,  WbV, WVb, (void*)Vt);

    flash_attn_kernel<<<dim3(TT / 64, BB * NHEADS), 128, 0, stream>>>(Qa, Ka, Vt, ctx);

    gemm_wmma_kernel<2><<<ggrid, 256, 0, stream>>>(ctx, WbO, WOb, d_out);
}
